// FallDetectionSTGCN_29068338659937
// MI455X (gfx1250) — compile-verified, matrix operand order checked by reading, not run on hardware
//
#include <hip/hip_runtime.h>
#include <hip/hip_bf16.h>

typedef _Float16 half_t;
typedef __attribute__((ext_vector_type(16))) _Float16 v16h;
typedef __attribute__((ext_vector_type(8)))  _Float16 v8h;
typedef __attribute__((ext_vector_type(8)))  float    v8f;

#define NJ   17
#define COUT 128
#define NC   (NJ * COUT)          // 2176
#define BN_RS 0.9999950000374997f // 1/sqrt(1 + 1e-5)

// ---------------------------------------------------------------------------
// WMMA fragment helpers (CDNA5 ISA 7.12.2 layouts, wave32)
// ---------------------------------------------------------------------------

__device__ inline v16h join16(v8h lo, v8h hi) {
  return __builtin_shufflevector(lo, hi, 0, 1, 2, 3, 4, 5, 6, 7,
                                 8, 9, 10, 11, 12, 13, 14, 15);
}

// A fragment: 16x32 f16 tile from row-major [rows x ld] half matrix.
// lane l: row = r0 + (l&15); K halves {k0+8*(l>>4)..+8} and {k0+16+8*(l>>4)..+8}
// -> two 16-byte vector loads per lane (16-B alignment guaranteed by layout).
__device__ inline v16h frag_a_global(const half_t* __restrict__ p, int ld,
                                     int r0, int k0, int rlo, int rhi) {
  int l   = threadIdx.x & 31;
  int row = r0 + (l & 15);
  int kb  = k0 + ((l >> 4) << 3);
  v8h lo, hi;
  if (row < rlo || row >= rhi) {
#pragma unroll
    for (int j = 0; j < 8; ++j) { lo[j] = (half_t)0.f; hi[j] = (half_t)0.f; }
  } else {
    const v8h* q = (const v8h*)(p + (long)row * ld + kb);
    lo = q[0];
    hi = q[2];   // +16 halves
  }
  return join16(lo, hi);
}

// B fragment from LDS pre-swizzled into fragment order:
// Bs[fragIdx*512 + lane*16 + j] -> one contiguous 32-byte load per lane.
__device__ inline v16h frag_b_swz(const half_t* __restrict__ s, int fragIdx) {
  int l = threadIdx.x & 31;
  const v16h* p = (const v16h*)(s + fragIdx * 512 + l * 16);
  return *p;
}

__device__ inline v8f vzero8() { v8f z = {0.f,0.f,0.f,0.f,0.f,0.f,0.f,0.f}; return z; }

__device__ inline v8f wmma_f16(const v16h& a, const v16h& b, const v8f& c) {
  return __builtin_amdgcn_wmma_f32_16x16x32_f16(false, a, false, b, (short)0, c,
                                                false, false);
}

// lane/half index -> K coordinate inside a 32-wide K tile
__device__ inline int swz_k(int l, int j) {
  return ((l >> 4) << 3) + (j & 7) + ((j >> 3) << 4);
}

// ---------------------------------------------------------------------------
// 0) build normalized adjacency  D^-1/2 (A+I) D^-1/2  (once, 289 floats)
// ---------------------------------------------------------------------------
__global__ void build_adj_kernel(float* __restrict__ adj) {
  if (threadIdx.x == 0 && blockIdx.x == 0) {
    float a[NJ][NJ];
    for (int i = 0; i < NJ; ++i)
      for (int j = 0; j < NJ; ++j) a[i][j] = (i == j) ? 1.f : 0.f;
    const int e[16][2] = {{0,1},{0,2},{1,3},{2,4},{5,7},{7,9},{6,8},{8,10},
                          {5,6},{5,11},{6,12},{11,13},{13,15},{12,14},{14,16},{11,12}};
    for (int k = 0; k < 16; ++k) {
      a[e[k][0]][e[k][1]] = 1.f;
      a[e[k][1]][e[k][0]] = 1.f;
    }
    float dinv[NJ];
    for (int i = 0; i < NJ; ++i) {
      float s = 0.f;
      for (int j = 0; j < NJ; ++j) s += a[i][j];
      dinv[i] = rsqrtf(s);
    }
    for (int i = 0; i < NJ; ++i)
      for (int j = 0; j < NJ; ++j) adj[i * NJ + j] = a[i][j] * dinv[i] * dinv[j];
  }
}

// ---------------------------------------------------------------------------
// 1) input transform: h0[m,c] = sum_i x[m,i]*w_in[c,i] + b_in[c]
// ---------------------------------------------------------------------------
__global__ void input_transform(const float* __restrict__ x,
                                const float* __restrict__ w_in,
                                const float* __restrict__ b_in,
                                half_t* __restrict__ h0, int Mtot) {
  __shared__ float w[64 * 3];
  __shared__ float bb[64];
  for (int i = threadIdx.x; i < 192; i += blockDim.x) w[i] = w_in[i];
  for (int i = threadIdx.x; i < 64;  i += blockDim.x) bb[i] = b_in[i];
  __syncthreads();
  int m = blockIdx.x * blockDim.x + threadIdx.x;
  if (m >= Mtot) return;
  float x0 = x[(long)m * 3 + 0], x1 = x[(long)m * 3 + 1], x2 = x[(long)m * 3 + 2];
  half_t* o = h0 + (long)m * 64;
#pragma unroll 4
  for (int c = 0; c < 64; ++c)
    o[c] = (half_t)(x0 * w[c * 3 + 0] + x1 * w[c * 3 + 1] + x2 * w[c * 3 + 2] + bb[c]);
}

// ---------------------------------------------------------------------------
// 2) support GEMM: S[m, o] = sum_k X[m,k] * gw[k,o]   (M x K) x (K x 128)
//    block = 256 thr (8 waves); M tile 128; weights pre-swizzled in LDS
// ---------------------------------------------------------------------------
__global__ void gemm_support(const half_t* __restrict__ X,
                             const float* __restrict__ gw,
                             half_t* __restrict__ S, int M, int K) {
  __shared__ half_t Bs[4 * 8 * 512];  // 32 KB max (K=128: KT=4, NT=8)
  const int KT = K >> 5;
  // stage + swizzle: Bs[((kt*8+nt)*32 + l)*16 + j] = gw[(kt*32+swz_k(l,j))*128 + nt*16+(l&15)]
  int tot = K * 128;
  for (int i = threadIdx.x; i < tot; i += blockDim.x) {
    int j  = i & 15;
    int l  = (i >> 4) & 31;
    int nt = (i >> 9) & 7;
    int kt = i >> 12;
    int kk  = (kt << 5) + swz_k(l, j);
    int col = (nt << 4) + (l & 15);
    Bs[i] = (half_t)gw[kk * 128 + col];
  }
  __syncthreads();

  int wave = threadIdx.x >> 5;
  int r0   = (blockIdx.x * 8 + wave) * 16;
  if (r0 >= M) return;

  v8f acc[8];
#pragma unroll
  for (int i = 0; i < 8; ++i) acc[i] = vzero8();

  for (int kt = 0; kt < KT; ++kt) {
    v16h a = frag_a_global(X, K, r0, kt << 5, 0, M);
#pragma unroll
    for (int nt = 0; nt < 8; ++nt) {
      v16h b  = frag_b_swz(Bs, kt * 8 + nt);
      acc[nt] = wmma_f16(a, b, acc[nt]);
    }
  }

  int l  = threadIdx.x & 31;
  int rb = r0 + ((l >> 4) << 3);
  int c0 = l & 15;
#pragma unroll
  for (int nt = 0; nt < 8; ++nt)
#pragma unroll
    for (int v = 0; v < 8; ++v)
      S[(long)(rb + v) * 128 + nt * 16 + c0] = (half_t)acc[nt][v];
}

// ---------------------------------------------------------------------------
// 3) adjacency mix + gb + BN1 + ReLU
// ---------------------------------------------------------------------------
__global__ void adj_bn_relu(const half_t* __restrict__ S,
                            const float* __restrict__ adjw,
                            const float* __restrict__ gb,
                            const float* __restrict__ bn1g,
                            const float* __restrict__ bn1b,
                            half_t* __restrict__ H) {
  __shared__ float  adj[NJ * NJ];
  __shared__ half_t Ss[NC];
  for (int i = threadIdx.x; i < NJ * NJ; i += blockDim.x) adj[i] = adjw[i];
  long bt = blockIdx.x;
  const half_t* Sp = S + bt * NC;
  for (int i = threadIdx.x; i < NC; i += blockDim.x) Ss[i] = Sp[i];
  __syncthreads();
  for (int i = threadIdx.x; i < NC; i += blockDim.x) {
    int m = i >> 7, o = i & 127;
    float acc = 0.f;
#pragma unroll
    for (int n = 0; n < NJ; ++n) acc += adj[m * NJ + n] * (float)Ss[n * 128 + o];
    acc += gb[o];
    acc = acc * (bn1g[i] * BN_RS) + bn1b[i];
    H[bt * NC + i] = (half_t)fmaxf(acc, 0.f);
  }
}

// ---------------------------------------------------------------------------
// 4) grouped (per-joint) temporal GEMM, taps in {1,3}:
//    out[t,co] = sum_k sum_ci W[n][co,ci,k] * Hin[b, t+k-taps/2, n, ci]
//    epilogue: (+bias)*bnScale+bnBias, then mode:
//      0: store (residual-branch conv, no relu)
//      1: relu, += Out[in-place]  (block0 residual pre-staged in Out)
//      2: relu, += res[identity]
//    grid = (B, 17, 2): z splits co into halves of 64.
// ---------------------------------------------------------------------------
__global__ void grouped_tconv_gemm(const half_t* __restrict__ Hin,
                                   const float* __restrict__ W,
                                   const float* __restrict__ bias,
                                   const float* __restrict__ bng,
                                   const float* __restrict__ bnb,
                                   const half_t* __restrict__ res,
                                   half_t* __restrict__ Out,
                                   int T, int Cin, int taps, int mode) {
  __shared__ half_t Ws[3 * 4 * 4 * 512];  // 48 KB max (taps=3, KT=4, NT=4)
  __shared__ float  epB[64], epG[64], epBB[64];
  const int b  = blockIdx.x;
  const int n  = blockIdx.y;
  const int gz = blockIdx.z;              // co half: [gz*64, gz*64+64)
  const int KT = Cin >> 5;

  // stage + swizzle weights: frag (k, kt, nt) at Ws[((k*KT+kt)*4+nt)*512 ...]
  int tot = taps * KT * 4 * 512;
  for (int i = threadIdx.x; i < tot; i += blockDim.x) {
    int j    = i & 15;
    int l    = (i >> 4) & 31;
    int nt   = (i >> 9) & 3;
    int rest = i >> 11;
    int kt   = rest % KT;
    int k    = rest / KT;
    int co   = gz * 64 + (nt << 4) + (l & 15);
    int ci   = (kt << 5) + swz_k(l, j);
    Ws[i] = (half_t)W[(((long)(n * 128 + co)) * Cin + ci) * taps + k];
  }
  for (int i = threadIdx.x; i < 64; i += blockDim.x) {
    int ch = n * 128 + gz * 64 + i;
    epB[i]  = bias[ch];
    epG[i]  = bng[ch] * BN_RS;
    epBB[i] = bnb[ch];
  }
  __syncthreads();

  const int ldH = NJ * Cin;
  const half_t* Hb = Hin + (long)b * T * ldH + n * Cin;
  half_t*       Ob = Out + (long)b * T * NC  + n * 128;
  const half_t* Rb = (mode == 2) ? (res + (long)b * T * NC + n * 128) : nullptr;

  int wave = threadIdx.x >> 5;
  int l    = threadIdx.x & 31;

  for (int mt = wave; mt < T / 16; mt += 8) {
    int t0 = mt * 16;
    v8f acc[4];
#pragma unroll
    for (int i = 0; i < 4; ++i) acc[i] = vzero8();

    for (int k = 0; k < taps; ++k) {
      int shift = k - (taps >> 1);
      for (int kt = 0; kt < KT; ++kt) {
        v16h a = frag_a_global(Hb, ldH, t0 + shift, kt << 5, 0, T);
        int fbase = (k * KT + kt) * 4;
#pragma unroll
        for (int nt = 0; nt < 4; ++nt) {
          v16h bf = frag_b_swz(Ws, fbase + nt);
          acc[nt] = wmma_f16(a, bf, acc[nt]);
        }
      }
    }

    int rb = t0 + ((l >> 4) << 3);
    int c0 = l & 15;
#pragma unroll
    for (int nt = 0; nt < 4; ++nt) {
      int cl = nt * 16 + c0;          // local co in [0,64)
      int co = gz * 64 + cl;          // co in [0,128)
#pragma unroll
      for (int v = 0; v < 8; ++v) {
        int   t   = rb + v;
        float val = (acc[nt][v] + epB[cl]) * epG[cl] + epBB[cl];
        long  oid = (long)t * NC + co;
        if (mode == 0) {
          Ob[oid] = (half_t)val;
        } else {
          val = fmaxf(val, 0.f);
          val += (float)((mode == 1) ? Ob[oid] : Rb[oid]);
          Ob[oid] = (half_t)val;
        }
      }
    }
  }
}

// ---------------------------------------------------------------------------
// 5) temporal mean
// ---------------------------------------------------------------------------
__global__ void temporal_mean(const half_t* __restrict__ H,
                              float* __restrict__ Mn, int B, int T) {
  int idx = blockIdx.x * blockDim.x + threadIdx.x;
  if (idx >= B * NC) return;
  int b = idx / NC, c = idx % NC;
  const half_t* p = H + (long)b * T * NC + c;
  float s = 0.f;
  for (int t = 0; t < T; ++t) s += (float)p[(long)t * NC];
  Mn[idx] = s * (1.f / (float)T);
}

// ---------------------------------------------------------------------------
// 6) MLP head: relu(M@w1+b1) -> relu(@w2+b2) -> sigmoid(@w3+b3); one block
// ---------------------------------------------------------------------------
__global__ void mlp_head(const float* __restrict__ Mn,
                         const float* __restrict__ w1, const float* __restrict__ b1,
                         const float* __restrict__ w2, const float* __restrict__ b2,
                         const float* __restrict__ w3, const float* __restrict__ b3,
                         float* __restrict__ out, int B) {
  __shared__ float h1[64 * 128];
  __shared__ float h2[64 * 64];
  for (int i = threadIdx.x; i < B * 128; i += blockDim.x) {
    int b = i >> 7, j = i & 127;
    float s = b1[j];
    const float* m = Mn + (long)b * NC;
    for (int k = 0; k < NC; ++k) s += m[k] * w1[k * 128 + j];
    h1[i] = fmaxf(s, 0.f);
  }
  __syncthreads();
  for (int i = threadIdx.x; i < B * 64; i += blockDim.x) {
    int b = i >> 6, j = i & 63;
    float s = b2[j];
#pragma unroll 8
    for (int k = 0; k < 128; ++k) s += h1[b * 128 + k] * w2[k * 64 + j];
    h2[i] = fmaxf(s, 0.f);
  }
  __syncthreads();
  for (int i = threadIdx.x; i < B; i += blockDim.x) {
    float s = b3[0];
#pragma unroll 8
    for (int k = 0; k < 64; ++k) s += h2[i * 64 + k] * w3[k];
    out[i] = 1.f / (1.f + expf(-s));
  }
}

// ---------------------------------------------------------------------------
// launcher
// ---------------------------------------------------------------------------
extern "C" void kernel_launch(void* const* d_in, const int* in_sizes, int n_in,
                              void* d_out, int out_size, void* d_ws, size_t ws_size,
                              hipStream_t stream) {
  (void)in_sizes; (void)n_in; (void)out_size;

  const int B = 64, T = 512;
  const long M = (long)B * T * NJ;            // 557056 rows

  const float* x    = (const float*)d_in[0];
  const float* w_in = (const float*)d_in[1];
  const float* b_in = (const float*)d_in[2];
  const float *gw[3], *gb[3], *bn1g[3], *bn1b[3], *cw[3], *cb[3], *bn2g[3], *bn2b[3];
  for (int i = 0; i < 3; ++i) {
    int base = 3 + 8 * i;
    gw[i]   = (const float*)d_in[base + 0];
    gb[i]   = (const float*)d_in[base + 1];
    bn1g[i] = (const float*)d_in[base + 2];
    bn1b[i] = (const float*)d_in[base + 3];
    cw[i]   = (const float*)d_in[base + 4];
    cb[i]   = (const float*)d_in[base + 5];
    bn2g[i] = (const float*)d_in[base + 6];
    bn2b[i] = (const float*)d_in[base + 7];
  }
  const float* rw0  = (const float*)d_in[27];
  const float* rb0  = (const float*)d_in[28];
  const float* rg0  = (const float*)d_in[29];
  const float* rbb0 = (const float*)d_in[30];
  const float* w1 = (const float*)d_in[31];
  const float* b1 = (const float*)d_in[32];
  const float* w2 = (const float*)d_in[33];
  const float* b2 = (const float*)d_in[34];
  const float* w3 = (const float*)d_in[35];
  const float* b3 = (const float*)d_in[36];
  float* out = (float*)d_out;

  // workspace: 3 ping-pong fp16 activation buffers + mean + adj
  const size_t CAPB = (size_t)M * 128 * sizeof(half_t);  // 142.6 MB
  char* ws = (char*)d_ws;
  half_t* P0  = (half_t*)(ws);
  half_t* P1  = (half_t*)(ws + CAPB);
  half_t* P2  = (half_t*)(ws + 2 * CAPB);
  float*  Mn  = (float*)(ws + 3 * CAPB);
  float*  adjw = Mn + (size_t)B * NC;
  if (ws_size < 3 * CAPB + ((size_t)B * NC + NJ * NJ) * sizeof(float)) return;

  build_adj_kernel<<<1, 32, 0, stream>>>(adjw);

  // input transform -> P0 [M, 64] f16
  input_transform<<<(int)((M + 255) / 256), 256, 0, stream>>>(x, w_in, b_in, P0, (int)M);

  dim3 convGrid(B, NJ, 2);
  int gemmBlocks = (int)(M / 128);   // exact: 4352

  // ---- block 0: in=P0 (Cin=64) ----
  gemm_support<<<gemmBlocks, 256, 0, stream>>>(P0, gw[0], P1, (int)M, 64);
  adj_bn_relu<<<B * T, 256, 0, stream>>>(P1, adjw, gb[0], bn1g[0], bn1b[0], P2);
  grouped_tconv_gemm<<<convGrid, 256, 0, stream>>>(P0, rw0, rb0, rg0, rbb0,
                                                   nullptr, P1, T, 64, 1, 0);
  grouped_tconv_gemm<<<convGrid, 256, 0, stream>>>(P2, cw[0], cb[0], bn2g[0], bn2b[0],
                                                   nullptr, P1, T, 128, 3, 1);

  // ---- block 1: in=P1 ----
  gemm_support<<<gemmBlocks, 256, 0, stream>>>(P1, gw[1], P2, (int)M, 128);
  adj_bn_relu<<<B * T, 256, 0, stream>>>(P2, adjw, gb[1], bn1g[1], bn1b[1], P0);
  grouped_tconv_gemm<<<convGrid, 256, 0, stream>>>(P0, cw[1], cb[1], bn2g[1], bn2b[1],
                                                   P1, P2, T, 128, 3, 2);

  // ---- block 2: in=P2 ----
  gemm_support<<<gemmBlocks, 256, 0, stream>>>(P2, gw[2], P0, (int)M, 128);
  adj_bn_relu<<<B * T, 256, 0, stream>>>(P0, adjw, gb[2], bn1g[2], bn1b[2], P1);
  grouped_tconv_gemm<<<convGrid, 256, 0, stream>>>(P1, cw[2], cb[2], bn2g[2], bn2b[2],
                                                   P2, P0, T, 128, 3, 2);

  // ---- head ----
  temporal_mean<<<(B * NC + 255) / 256, 256, 0, stream>>>(P0, Mn, B, T);
  mlp_head<<<1, 256, 0, stream>>>(Mn, w1, b1, w2, b2, w3, b3, out, B);
}
